// MLPBlock_86096914416234
// MI455X (gfx1250) — matrix operand order repaired and checked by reading, not compiled, not run on hardware
//
#include <hip/hip_runtime.h>
#include <hip/hip_bf16.h>

// ---------------- problem sizes ----------------
#define TOK   2048     // B*S
#define HDIM  2880
#define IDIM  2880
#define TWOI  5760
#define NEXP  16
#define TOPK  4

// ---------------- GEMM tiling ------------------
#define MT  128
#define NT  64
#define KT  64
#define LDA 72         // padded bf16 row stride in LDS (avoid bank conflicts)

typedef __attribute__((ext_vector_type(16))) __bf16 bf16x16;
typedef __attribute__((ext_vector_type(2)))  __bf16 bf16x2;
typedef __attribute__((ext_vector_type(8)))  float  f32x8;
typedef __attribute__((ext_vector_type(4)))  unsigned int uint32x4;
typedef __attribute__((ext_vector_type(4)))  int    int32x4;
typedef __attribute__((ext_vector_type(8)))  int    int32x8;

#if defined(__AMDGCN__) && __has_builtin(__builtin_amdgcn_tensor_load_to_lds)
#define HAVE_TDM 1
#else
#define HAVE_TDM 0
#endif

#if defined(__AMDGCN__) && __has_builtin(__builtin_amdgcn_global_load_async_to_lds_b128)
#define HAVE_ASYNC 1
#else
#define HAVE_ASYNC 0
#endif

#if HAVE_ASYNC
typedef __attribute__((address_space(1))) int32x4 gas_int4;   // global
typedef __attribute__((address_space(3))) int32x4 las_int4;   // LDS

static __device__ __forceinline__ void async_cp16(const unsigned short* g,
                                                  unsigned short* d) {
  __builtin_amdgcn_global_load_async_to_lds_b128(
      (gas_int4*)g, (las_int4*)d, 0, 0);
}

static __device__ __forceinline__ void wait_async0() {
#if __has_builtin(__builtin_amdgcn_s_wait_asynccnt)
  __builtin_amdgcn_s_wait_asynccnt(0);
#else
  asm volatile("s_wait_asynccnt 0x0" ::: "memory");
#endif
}
#endif

struct U32x8 { uint4 lo, hi; };

static __device__ __forceinline__ unsigned short f2bf(float f) {
  unsigned int u = __builtin_bit_cast(unsigned int, f);
  u += 0x7fffu + ((u >> 16) & 1u);      // round-to-nearest-even
  return (unsigned short)(u >> 16);
}

// two f32 -> packed bf16 pair: round-half-up then byte-select (1x v_perm_b32)
static __device__ __forceinline__ unsigned pk2bf(float a, float b) {
#if defined(__AMDGCN__) && __has_builtin(__builtin_amdgcn_cvt_pk_bf16_f32)
  bf16x2 p = __builtin_amdgcn_cvt_pk_bf16_f32(a, b);
  return __builtin_bit_cast(unsigned, p);
#elif defined(__AMDGCN__)
  unsigned ua = __builtin_bit_cast(unsigned, a) + 0x8000u;
  unsigned ub = __builtin_bit_cast(unsigned, b) + 0x8000u;
  // out = { ub[31:16], ua[31:16] }
  return __builtin_amdgcn_perm(ub, ua, 0x07060302u);
#else
  return (unsigned)f2bf(a) | ((unsigned)f2bf(b) << 16);
#endif
}

static __device__ __forceinline__ unsigned long long pk4bf(float4 v) {
  unsigned lo = pk2bf(v.x, v.y);
  unsigned hi = pk2bf(v.z, v.w);
  return (unsigned long long)lo | ((unsigned long long)hi << 32);
}

static __device__ __forceinline__ bf16x16 ld_frag2(const unsigned short* p0,
                                                   const unsigned short* p1) {
  U32x8 t;
  t.lo = *(const uint4*)p0;             // ds_load_b128
  t.hi = *(const uint4*)p1;             // ds_load_b128
  return __builtin_bit_cast(bf16x16, t);
}

// ---------------- init: zero output + counters ----------------
__global__ __launch_bounds__(256) void k_init(float* __restrict__ out, int* __restrict__ cnt) {
  long i4 = ((long)blockIdx.x * 256 + threadIdx.x) * 4;
  if (i4 + 3 < (long)TOK * HDIM) {
    float4 z = make_float4(0.f, 0.f, 0.f, 0.f);
    *(float4*)(out + i4) = z;
  }
  if (blockIdx.x == 0 && threadIdx.x < NEXP) cnt[threadIdx.x] = 0;
}

// ---------------- x: fp32 -> bf16 ----------------
__global__ __launch_bounds__(256) void k_cvt(const float* __restrict__ x,
                                             unsigned short* __restrict__ xb) {
  long i4 = ((long)blockIdx.x * 256 + threadIdx.x) * 4;
  if (i4 + 3 < (long)TOK * HDIM) {
    float4 v = *(const float4*)(x + i4);
    *(unsigned long long*)(xb + i4) = pk4bf(v);
  }
}

// ---------------- router: logits -> top4 -> softmax -> slots ----------------
__global__ __launch_bounds__(128) void k_router(const float* __restrict__ x,
                                                const float* __restrict__ gw,
                                                const float* __restrict__ gb,
                                                int* __restrict__ cnt,
                                                int* __restrict__ t4e,
                                                int* __restrict__ t4s,
                                                float* __restrict__ t4w) {
  __shared__ float sx[HDIM];
  __shared__ float slog[NEXP];
  int t = blockIdx.x;
  const float* xr = x + (size_t)t * HDIM;
  for (int i = threadIdx.x; i < HDIM; i += 128) sx[i] = xr[i];
  __syncthreads();

  int wave = threadIdx.x >> 5, lane = threadIdx.x & 31;
  #pragma unroll
  for (int j = 0; j < 4; ++j) {
    int e = wave * 4 + j;
    const float* w = gw + (size_t)e * HDIM;
    float s = 0.f;
    for (int k = lane; k < HDIM; k += 32) s += sx[k] * w[k];
    #pragma unroll
    for (int off = 16; off > 0; off >>= 1) s += __shfl_xor(s, off, 32);
    if (lane == 0) slog[e] = s + gb[e];
  }
  __syncthreads();

  if (threadIdx.x == 0) {
    float lg[NEXP];
    #pragma unroll
    for (int e = 0; e < NEXP; ++e) lg[e] = slog[e];
    int idx[TOPK]; float val[TOPK];
    unsigned mask = 0;
    #pragma unroll
    for (int k = 0; k < TOPK; ++k) {
      int bi = 0; float bv = -3.4e38f;
      for (int e = 0; e < NEXP; ++e)
        if (!((mask >> e) & 1u) && lg[e] > bv) { bv = lg[e]; bi = e; }
      mask |= 1u << bi; idx[k] = bi; val[k] = bv;
    }
    float m = val[0], ssum = 0.f, w4[TOPK];
    #pragma unroll
    for (int k = 0; k < TOPK; ++k) { w4[k] = __expf(val[k] - m); ssum += w4[k]; }
    float inv = 1.f / ssum;
    #pragma unroll
    for (int k = 0; k < TOPK; ++k) {
      int e = idx[k];
      int s = atomicAdd(&cnt[e], 1);
      t4e[t * TOPK + k] = e;
      t4s[t * TOPK + k] = s;
      t4w[t * TOPK + k] = w4[k] * inv;
    }
  }
}

// ---------------- exclusive prefix over 16 counts ----------------
__global__ void k_prefix(const int* __restrict__ cnt, int* __restrict__ basep) {
  if (threadIdx.x == 0 && blockIdx.x == 0) {
    int acc = 0;
    for (int e = 0; e < NEXP; ++e) { basep[e] = acc; acc += cnt[e]; }
  }
}

// ---------------- build compacted row lists ----------------
__global__ __launch_bounds__(256) void k_build(const int* __restrict__ basep,
                                               const int* __restrict__ t4e,
                                               const int* __restrict__ t4s,
                                               const float* __restrict__ t4w,
                                               int* __restrict__ rowTok,
                                               float* __restrict__ rowW) {
  int i = blockIdx.x * 256 + threadIdx.x;
  if (i < TOK * TOPK) {
    int e = t4e[i];
    int r = basep[e] + t4s[i];
    rowTok[r] = i / TOPK;
    rowW[r]   = t4w[i];
  }
}

// ---------------- gate_up GEMM + SwiGLU (bf16 WMMA, double-buffered) ----------------
__global__ __launch_bounds__(256, 1) void k_gateup(
    const unsigned short* __restrict__ xb,
    const float* __restrict__ wgu, const float* __restrict__ bgu,
    const float* __restrict__ alpha, const float* __restrict__ beta,
    const float* __restrict__ lim,
    const int* __restrict__ cnt, const int* __restrict__ basep,
    const int* __restrict__ rowTok,
    unsigned short* __restrict__ act) {
  __shared__ unsigned short lA [2][MT][LDA];
  __shared__ unsigned short lBg[2][NT][LDA];
  __shared__ unsigned short lBl[2][NT][LDA];

  int e = blockIdx.z;
  int nrows_e = cnt[e];
  int m0 = blockIdx.y * MT;
  if (m0 >= nrows_e) return;
  int rbase = basep[e];
  int n0 = blockIdx.x * NT;
  int nrows = min(MT, nrows_e - m0);

  int tid = threadIdx.x;
  int wv = tid >> 5, lane = tid & 31;
  int wm = wv >> 1, wn = wv & 1;

  // A staging: 2 threads per row, 32 bf16 (64B) each; clamp OOB rows to a
  // valid row (their outputs are discarded in the epilogue).
  int am = tid >> 1, ah = tid & 1;
  int arow = (am < nrows) ? (rbase + m0 + am) : (rbase + m0);
  const unsigned short* asrc = xb + (size_t)rowTok[arow] * HDIM + ah * 32;

  // B staging: 128 logical rows (64 glu + 64 lin), 2 threads per row
  int blr = tid >> 1, bth = tid & 1;
  int wrow = (blr < NT) ? (n0 + blr) : (IDIM + n0 + (blr - NT));
  const float* bsrc = wgu + (size_t)e * TWOI * HDIM + (size_t)wrow * HDIM + bth * 32;

#if HAVE_ASYNC
  auto asyncA = [&](int kt, int buf) {
    const unsigned short* g = asrc + kt;
    unsigned short* d = &lA[buf][am][ah * 32];
    #pragma unroll
    for (int i = 0; i < 4; ++i)
      async_cp16(g + i * 8, d + i * 8);
  };
#else
  uint4 aR[4];
  auto loadA = [&](int kt) {
    const uint4* s = (const uint4*)(asrc + kt);
    aR[0] = s[0]; aR[1] = s[1]; aR[2] = s[2]; aR[3] = s[3];
  };
  auto storeA = [&](int buf) {
    unsigned short* d = &lA[buf][am][ah * 32];
    *(uint4*)(d +  0) = aR[0]; *(uint4*)(d +  8) = aR[1];
    *(uint4*)(d + 16) = aR[2]; *(uint4*)(d + 24) = aR[3];
  };
#endif

  float4 bR[8];
  auto loadB = [&](int kt) {
    const float* s = bsrc + kt;
    #pragma unroll
    for (int i = 0; i < 8; ++i) bR[i] = *(const float4*)(s + i * 4);
  };
  auto storeB = [&](int buf) {
    unsigned short* d = (blr < NT) ? &lBg[buf][blr][bth * 32]
                                   : &lBl[buf][blr - NT][bth * 32];
    #pragma unroll
    for (int i = 0; i < 8; ++i)
      *(unsigned long long*)(d + i * 4) = pk4bf(bR[i]);
  };

  f32x8 zero8 = {0.f,0.f,0.f,0.f,0.f,0.f,0.f,0.f};
  f32x8 cg[2][2], cl[2][2];
  #pragma unroll
  for (int i = 0; i < 2; ++i)
    #pragma unroll
    for (int j = 0; j < 2; ++j) { cg[i][j] = zero8; cl[i][j] = zero8; }

  auto compute = [&](int buf) {
    #pragma unroll
    for (int kk = 0; kk < KT; kk += 32) {
      int khalf = (lane >> 4) * 8;          // A: lanes 16-31 hold K+8
      int kb    = kk + (lane >> 4) * 8;     // B: lanes 16-31 hold K+8
      bf16x16 af[2], bgf[2], blf[2];
      #pragma unroll
      for (int mi = 0; mi < 2; ++mi) {
        int mr = wm * 32 + mi * 16 + (lane & 15);
        af[mi] = ld_frag2(&lA[buf][mr][kk + khalf], &lA[buf][mr][kk + khalf + 16]);
      }
      #pragma unroll
      for (int ni = 0; ni < 2; ++ni) {
        int nc = wn * 32 + ni * 16 + (lane & 15);
        bgf[ni] = ld_frag2(&lBg[buf][nc][kb], &lBg[buf][nc][kb + 16]);
        blf[ni] = ld_frag2(&lBl[buf][nc][kb], &lBl[buf][nc][kb + 16]);
      }
      #pragma unroll
      for (int mi = 0; mi < 2; ++mi)
        #pragma unroll
        for (int ni = 0; ni < 2; ++ni) {
          cg[mi][ni] = __builtin_amdgcn_wmma_f32_16x16x32_bf16(
              false, af[mi], false, bgf[ni], (short)0, cg[mi][ni], false, false);
          cl[mi][ni] = __builtin_amdgcn_wmma_f32_16x16x32_bf16(
              false, af[mi], false, blf[ni], (short)0, cl[mi][ni], false, false);
        }
    }
  };

  // ---- double-buffered main loop ----
#if HAVE_ASYNC
  asyncA(0, 0);
#else
  loadA(0); storeA(0);
#endif
  loadB(0); storeB(0);
#if HAVE_ASYNC
  wait_async0();
#endif
  __syncthreads();

  const int NS = HDIM / KT;
  for (int s = 0; s < NS; ++s) {
    int nxt = (s + 1 < NS) ? (s + 1) : s;
    int nb  = (s + 1) & 1;
#if HAVE_ASYNC
    if (s + 1 < NS) asyncA((s + 1) * KT, nb);  // DMA overlaps the WMMAs
#else
    loadA(nxt * KT);
#endif
    loadB(nxt * KT);                           // global loads issue early
    compute(s & 1);
#if !HAVE_ASYNC
    storeA(nb);
#endif
    storeB(nb);
#if HAVE_ASYNC
    wait_async0();
#endif
    __syncthreads();
  }

  // SwiGLU epilogue
  float al = alpha[e], be = beta[e], lm = lim[e];
  float bgv[2], blv[2]; int ncol[2];
  #pragma unroll
  for (int ni = 0; ni < 2; ++ni) {
    int n = n0 + wn * 32 + ni * 16 + (lane & 15);
    ncol[ni] = n;
    bgv[ni] = bgu[(size_t)e * TWOI + n];
    blv[ni] = bgu[(size_t)e * TWOI + IDIM + n];
  }
  #pragma unroll
  for (int mi = 0; mi < 2; ++mi) {
    int rl0 = wm * 32 + mi * 16 + ((lane >> 4) * 8);
    #pragma unroll
    for (int i = 0; i < 8; ++i) {
      int rl = rl0 + i;
      if (rl < nrows) {
        size_t rowoff = (size_t)(rbase + m0 + rl) * IDIM;
        #pragma unroll
        for (int ni = 0; ni < 2; ++ni) {
          float g = cg[mi][ni][i] + bgv[ni];
          float l = cl[mi][ni][i] + blv[ni];
          g = fminf(g, lm);
          l = fminf(fmaxf(l, -lm), lm);
          float a = g * (1.f / (1.f + __expf(-al * g))) * (l + be);
          act[rowoff + ncol[ni]] = f2bf(a);
        }
      }
    }
  }
}

// ---------------- down GEMM + weighted combine (TDM A-tile, double-buffered) ----------------
__global__ __launch_bounds__(256, 1) void k_down(
    const unsigned short* __restrict__ act,
    const float* __restrict__ wd, const float* __restrict__ bd,
    const int* __restrict__ cnt, const int* __restrict__ basep,
    const int* __restrict__ rowTok, const float* __restrict__ rowW,
    float* __restrict__ out) {
  __shared__ unsigned short lA[2][MT][LDA];
  __shared__ unsigned short lB[2][NT][LDA];

  int e = blockIdx.z;
  int nrows_e = cnt[e];
  int m0 = blockIdx.y * MT;
  if (m0 >= nrows_e) return;
  int rbase = basep[e];
  int n0 = blockIdx.x * NT;
  int nrows = min(MT, nrows_e - m0);

  int tid = threadIdx.x;
  int wv = tid >> 5, lane = tid & 31;
  int wm = wv >> 1, wn = wv & 1;

  // B staging: 64 rows x 64 k fp32 -> 4 threads per row, 16 floats each
  int blr = tid >> 2, btq = tid & 3;
  const float* bsrc = wd + ((size_t)e * HDIM + (size_t)(n0 + blr)) * IDIM + btq * 16;
  float4 bR[4];
  auto loadB = [&](int kt) {
    const float* s = bsrc + kt;
    #pragma unroll
    for (int i = 0; i < 4; ++i) bR[i] = *(const float4*)(s + i * 4);
  };
  auto storeB = [&](int buf) {
    unsigned short* d = &lB[buf][blr][btq * 16];
    #pragma unroll
    for (int i = 0; i < 4; ++i)
      *(unsigned long long*)(d + i * 4) = pk4bf(bR[i]);
  };

#if HAVE_TDM
  // A tile via Tensor Data Mover: 2D tile 64 elems x 128 rows of bf16,
  // stride = IDIM, hardware pad 4 DWORDs every 32 DWORDs -> LDS stride 72 bf16.
  const unsigned short* abase = act + (size_t)(rbase + m0) * IDIM;
  auto tdmA = [&](int kt, int buf) {
    if (wv == 0) {
      unsigned long long ga = (unsigned long long)(uintptr_t)(abase + kt);
      unsigned ldsb = (unsigned)(uintptr_t)(&lA[buf][0][0]);
      uint32x4 g0 = { 1u,                                   // count=1
                      ldsb,                                 // lds_addr
                      (unsigned)(ga & 0xffffffffu),         // global_addr lo
                      (unsigned)((ga >> 32) & 0x01ffffffu) | 0x80000000u }; // hi + type=2
      int32x8 g1 = { (int)((1u << 16)   // data_size = 2B
                         | (1u << 20)   // pad_enable
                         | (4u << 22)   // pad_interval: every 32 DWORDs
                         | (3u << 25)), // pad_amount: 4 DWORDs
                     (int)(IDIM << 16),            // tensor_dim0 lo16 @ bits 63:48
                     0,                            // tensor_dim0 hi16 / dim1 lo16
                     (int)(0x10u | ((unsigned)KT << 16)), // dim1 hi16 (1M) | tile_dim0=64
                     MT,                           // tile_dim1=128, tile_dim2=0
                     IDIM,                         // tensor_dim0_stride lo32
                     0, 0 };
      int32x4 z4 = {0, 0, 0, 0};
#if defined(__clang_major__) && __clang_major__ >= 23
      int32x8 z8 = {0, 0, 0, 0, 0, 0, 0, 0};
      __builtin_amdgcn_tensor_load_to_lds(g0, g1, z4, z4, z8, 0);
#else
      __builtin_amdgcn_tensor_load_to_lds(g0, g1, z4, z4, 0);
#endif
    }
  };
#else
  int am = tid >> 1, ah = tid & 1;
  // act rows are allocated with MT rows of padding, so no clamp needed
  const unsigned short* asrc = act + (size_t)(rbase + m0 + am) * IDIM + ah * 32;
#if HAVE_ASYNC
  auto asyncA = [&](int kt, int buf) {
    const unsigned short* g = asrc + kt;
    unsigned short* d = &lA[buf][am][ah * 32];
    #pragma unroll
    for (int i = 0; i < 4; ++i)
      async_cp16(g + i * 8, d + i * 8);
  };
#else
  uint4 aR[4];
  auto loadA = [&](int kt) {
    const uint4* s = (const uint4*)(asrc + kt);
    aR[0] = s[0]; aR[1] = s[1]; aR[2] = s[2]; aR[3] = s[3];
  };
  auto storeA = [&](int buf) {
    unsigned short* d = &lA[buf][am][ah * 32];
    *(uint4*)(d +  0) = aR[0]; *(uint4*)(d +  8) = aR[1];
    *(uint4*)(d + 16) = aR[2]; *(uint4*)(d + 24) = aR[3];
  };
#endif
#endif

  f32x8 zero8 = {0.f,0.f,0.f,0.f,0.f,0.f,0.f,0.f};
  f32x8 c[2][2];
  #pragma unroll
  for (int i = 0; i < 2; ++i)
    #pragma unroll
    for (int j = 0; j < 2; ++j) c[i][j] = zero8;

  auto compute = [&](int buf) {
    #pragma unroll
    for (int kk = 0; kk < KT; kk += 32) {
      int khalf = (lane >> 4) * 8;
      int kb    = kk + (lane >> 4) * 8;
      bf16x16 af[2], bf[2];
      #pragma unroll
      for (int mi = 0; mi < 2; ++mi) {
        int mr = wm * 32 + mi * 16 + (lane & 15);
        af[mi] = ld_frag2(&lA[buf][mr][kk + khalf], &lA[buf][mr][kk + khalf + 16]);
      }
      #pragma unroll
      for (int ni = 0; ni < 2; ++ni) {
        int nc = wn * 32 + ni * 16 + (lane & 15);
        bf[ni] = ld_frag2(&lB[buf][nc][kb], &lB[buf][nc][kb + 16]);
      }
      #pragma unroll
      for (int mi = 0; mi < 2; ++mi)
        #pragma unroll
        for (int ni = 0; ni < 2; ++ni)
          c[mi][ni] = __builtin_amdgcn_wmma_f32_16x16x32_bf16(
              false, af[mi], false, bf[ni], (short)0, c[mi][ni], false, false);
    }
  };

  // ---- double-buffered main loop ----
#if HAVE_TDM
  tdmA(0, 0);
#elif HAVE_ASYNC
  asyncA(0, 0);
#else
  loadA(0); storeA(0);
#endif
  loadB(0); storeB(0);
#if HAVE_TDM
  __builtin_amdgcn_s_wait_tensorcnt(0);
#elif HAVE_ASYNC
  wait_async0();
#endif
  __syncthreads();

  const int NS = IDIM / KT;
  for (int s = 0; s < NS; ++s) {
    int nxt = (s + 1 < NS) ? (s + 1) : s;
    int nb  = (s + 1) & 1;
#if HAVE_TDM
    if (s + 1 < NS) tdmA((s + 1) * KT, nb);   // async DMA overlaps WMMAs
#elif HAVE_ASYNC
    if (s + 1 < NS) asyncA((s + 1) * KT, nb);
#else
    loadA(nxt * KT);
#endif
    loadB(nxt * KT);
    compute(s & 1);
#if !HAVE_TDM && !HAVE_ASYNC
    storeA(nb);
#endif
    storeB(nb);
#if HAVE_TDM
    __builtin_amdgcn_s_wait_tensorcnt(0);
#elif HAVE_ASYNC
    wait_async0();
#endif
    __syncthreads();
  }

  // epilogue: out[token] += w * (y + bd)
  int hcol[2]; float bdv[2];
  #pragma unroll
  for (int ni = 0; ni < 2; ++ni) {
    hcol[ni] = n0 + wn * 32 + ni * 16 + (lane & 15);
    bdv[ni]  = bd[(size_t)e * HDIM + hcol[ni]];
  }
  #pragma unroll
  for (int mi = 0; mi < 2; ++mi) {
    int rl0 = wm * 32 + mi * 16 + ((lane >> 4) * 8);
    #pragma unroll
    for (int i = 0; i < 8; ++i) {
      int rl = rl0 + i;
      if (rl < nrows) {
        int row = rbase + m0 + rl;
        int tok = rowTok[row];
        float wgt = rowW[row];
        float* orow = out + (size_t)tok * HDIM;
        #pragma unroll
        for (int ni = 0; ni < 2; ++ni)
          atomicAdd(orow + hcol[ni], (c[mi][ni][i] + bdv[ni]) * wgt);
      }
    }
  }
}

// ---------------- host launcher ----------------
extern "C" void kernel_launch(void* const* d_in, const int* in_sizes, int n_in,
                              void* d_out, int out_size, void* d_ws, size_t ws_size,
                              hipStream_t stream) {
  (void)in_sizes; (void)n_in; (void)out_size; (void)ws_size;
  const float* x     = (const float*)d_in[0];
  const float* gw    = (const float*)d_in[1];
  const float* gb    = (const float*)d_in[2];
  const float* wgu   = (const float*)d_in[3];
  const float* bgu   = (const float*)d_in[4];
  const float* wd    = (const float*)d_in[5];
  const float* bd    = (const float*)d_in[6];
  const float* alpha = (const float*)d_in[7];
  const float* beta  = (const float*)d_in[8];
  const float* lim   = (const float*)d_in[9];
  float* out = (float*)d_out;

  char* ws = (char*)d_ws;
  size_t off = 0;
  auto take = [&](size_t bytes) { char* p = ws + off; off = (off + bytes + 255) & ~(size_t)255; return p; };
  unsigned short* xb   = (unsigned short*)take((size_t)TOK * HDIM * 2);
  // +MT row pad: fixed-size 128-row TDM tiles may read past the last row
  unsigned short* actb = (unsigned short*)take(((size_t)TOK * TOPK + MT) * IDIM * 2);
  int*   cnt    = (int*)take(NEXP * 4);
  int*   basep  = (int*)take(NEXP * 4);
  int*   rowTok = (int*)take((size_t)TOK * TOPK * 4);
  float* rowW   = (float*)take((size_t)TOK * TOPK * 4);
  int*   t4e    = (int*)take((size_t)TOK * TOPK * 4);
  int*   t4s    = (int*)take((size_t)TOK * TOPK * 4);
  float* t4w    = (float*)take((size_t)TOK * TOPK * 4);

  long nelem = (long)TOK * HDIM;
  int blocks4 = (int)((nelem / 4 + 255) / 256);

  k_init  <<<blocks4, 256, 0, stream>>>(out, cnt);
  k_cvt   <<<blocks4, 256, 0, stream>>>(x, xb);
  k_router<<<TOK, 128, 0, stream>>>(x, gw, gb, cnt, t4e, t4s, t4w);
  k_prefix<<<1, 32, 0, stream>>>(cnt, basep);
  k_build <<<(TOK * TOPK + 255) / 256, 256, 0, stream>>>(basep, t4e, t4s, t4w, rowTok, rowW);

  dim3 g1(IDIM / NT, TOK / MT, NEXP);   // 45 x 16 x 16
  k_gateup<<<g1, 256, 0, stream>>>(xb, wgu, bgu, alpha, beta, lim,
                                   cnt, basep, rowTok, actb);
  dim3 g2(HDIM / NT, TOK / MT, NEXP);   // 45 x 16 x 16
  k_down  <<<g2, 256, 0, stream>>>(actb, wd, bd, cnt, basep, rowTok, rowW, out);
}
